// VD_Conv_67534065762904
// MI455X (gfx1250) — matrix operand order repaired and checked by reading, not compiled, not run on hardware
//
#include <hip/hip_runtime.h>

// ---------------------------------------------------------------------------
// VD graph-conv block for MI455X (gfx1250, wave32, WMMA).
// All GEMMs run as v_wmma_f32_16x16x32_bf16 (bf16 inputs, fp32 accum).
// Edge ef-projection GEMM is fused with gather/scatter (atomic f32 adds),
// with gathers batched 8-at-a-time for memory-level parallelism and
// 32-bit saddr-form addressing (SGPR base + voffset, scale_offset).
// ---------------------------------------------------------------------------

typedef __attribute__((ext_vector_type(16))) __bf16 v16bf;
typedef __attribute__((ext_vector_type(8)))  float  v8f;

constexpr int Bc = 4;
constexpr int Nn = 10000;
constexpr int Ee = 160000;
constexpr int H  = 128;
constexpr int EF = 32;
constexpr int MT = Bc * Nn;          // 40000 node rows total
constexpr int RPW = 64;              // rows per workgroup in GEMM

union AFrag { uint4 u4[2]; v16bf v; };
union H8    { __bf16 h[8]; uint4 u; };

// ---------------------------------------------------------------------------
// Weight prep: bf16-convert + transpose to [out_col][k] so WMMA B fragments
// (two runs of 8 consecutive K for a fixed column) are contiguous 16B reads.
// wt holds 10 HxH matrices: emb00,emb01,emb10,emb11,conv00,conv01,conv10,
// conv11,out0,out1. ewt holds e_w transposed to [H][EF].
// ---------------------------------------------------------------------------
__global__ __launch_bounds__(256) void k_convert_weights(
    const float* __restrict__ emb_w, const float* __restrict__ conv_w,
    const float* __restrict__ out_w, const float* __restrict__ e_w,
    __bf16* __restrict__ wt, __bf16* __restrict__ ewt)
{
  int g = blockIdx.x * 256 + threadIdx.x;
  if (g < 10 * H * H) {
    int m = g / (H * H);
    int rem = g - m * (H * H);
    int col = rem / H, k = rem - col * H;
    const float* src = (m < 4) ? (emb_w + m * H * H)
                     : (m < 8) ? (conv_w + (m - 4) * H * H)
                               : (out_w + (m - 8) * H * H);
    wt[g] = (__bf16)src[k * H + col];
  } else {
    int i = g - 10 * H * H;
    if (i < H * EF) {
      int col = i / EF, k = i - col * EF;
      ewt[i] = (__bf16)e_w[k * H + col];
    }
  }
}

// ---------------------------------------------------------------------------
// Tiled GEMM + SiLU (+ optional residual add, per-batch C scale, elementwise
// multiply). Y[r, c] = epi( silu(A[r,:] @ W[:,c] + bias[c]) ).
// WG = 256 threads = 8 waves; WG tile = 64 rows x 128 cols; wave tile =
// 64 rows x 16 cols as 4 accumulators; K-loop 128/32 = 4 WMMAs each.
// ---------------------------------------------------------------------------
template<bool A_F32, bool SCALE_A, bool OUT_B, bool OUT_F,
         bool RES, bool SCALE_RES, bool MUL>
__global__ __launch_bounds__(256) void k_gemm_silu(
    const void* __restrict__ Ap,            // [MT,H] bf16 or f32
    const __bf16* __restrict__ Wt,          // [H][H] transposed, bf16
    const float* __restrict__ bias,         // [H]
    const float* __restrict__ Cv,           // [B] per-batch scale
    const float* __restrict__ resid,        // [MT,H] f32
    const float* __restrict__ mulsrc,       // [MT,H] f32
    float* __restrict__ outF,               // [MT,H] f32
    __bf16* __restrict__ outB)              // [MT,H] bf16
{
  constexpr int SA = 136;   // A LDS row stride (bf16): 272B, bank-spread
  constexpr int SW = 144;   // W LDS row stride (bf16): 288B, bank-spread
  __shared__ __align__(16) __bf16 sA[RPW * SA];
  __shared__ __align__(16) __bf16 sW[H * SW];

  const int tid = threadIdx.x;
  const int r0 = blockIdx.x * RPW;

  // Stage W: 128x128 bf16 = 2048 x 16B chunks, 8 per thread.
  {
    const uint4* src = (const uint4*)Wt;
#pragma unroll
    for (int i = 0; i < 8; ++i) {
      int c = tid + i * 256;
      int row = c >> 4, seg = c & 15;
      *(uint4*)&sW[row * SW + seg * 8] = src[c];
    }
  }
  // Stage A: 64x128, as bf16 (copy) or f32 (convert, optional *C[b]).
  if (A_F32) {
    const float4* src = (const float4*)Ap;
#pragma unroll
    for (int i = 0; i < 4; ++i) {
      int c = tid + i * 256;
      int row = c >> 4, seg = c & 15;
      int grow = r0 + row;
      float4 a = src[grow * 32 + seg * 2];
      float4 b = src[grow * 32 + seg * 2 + 1];
      if (SCALE_A) {
        float s = Cv[grow / Nn];
        a.x *= s; a.y *= s; a.z *= s; a.w *= s;
        b.x *= s; b.y *= s; b.z *= s; b.w *= s;
      }
      H8 p;
      p.h[0] = (__bf16)a.x; p.h[1] = (__bf16)a.y;
      p.h[2] = (__bf16)a.z; p.h[3] = (__bf16)a.w;
      p.h[4] = (__bf16)b.x; p.h[5] = (__bf16)b.y;
      p.h[6] = (__bf16)b.z; p.h[7] = (__bf16)b.w;
      *(uint4*)&sA[row * SA + seg * 8] = p.u;
    }
  } else {
    const uint4* src = (const uint4*)Ap;
#pragma unroll
    for (int i = 0; i < 4; ++i) {
      int c = tid + i * 256;
      int row = c >> 4, seg = c & 15;
      *(uint4*)&sA[row * SA + seg * 8] = src[(r0 + row) * 16 + seg];
    }
  }
  __syncthreads();

  const int lane = tid & 31;
  const int wv = tid >> 5;
  const int lm = lane & 15;
  const int hi = lane >> 4;

  v8f z = {0.f, 0.f, 0.f, 0.f, 0.f, 0.f, 0.f, 0.f};
  v8f acc[4];
#pragma unroll
  for (int t = 0; t < 4; ++t) acc[t] = z;

#pragma unroll
  for (int kk = 0; kk < 4; ++kk) {
    // B fragment: column N = 16*wv + lm; K chunks {kk*32+hi*8, kk*32+16+hi*8}.
    AFrag bf;
    const int nrow = wv * 16 + lm;
    const int kof = kk * 32 + hi * 8;
    bf.u4[0] = *(const uint4*)&sW[nrow * SW + kof];
    bf.u4[1] = *(const uint4*)&sW[nrow * SW + kof + 16];
#pragma unroll
    for (int t = 0; t < 4; ++t) {
      // A fragment: row M = t*16 + lm; K chunks {kk*32+hi*8, kk*32+16+hi*8}.
      AFrag af;
      const int arow = t * 16 + lm;
      const int ak = kk * 32 + hi * 8;
      af.u4[0] = *(const uint4*)&sA[arow * SA + ak];
      af.u4[1] = *(const uint4*)&sA[arow * SA + ak + 16];
      acc[t] = __builtin_amdgcn_wmma_f32_16x16x32_bf16(
          false, af.v, false, bf.v, (short)0, acc[t], false, false);
    }
  }

  // Epilogue: bias + SiLU (+ residual, + mul), scatter by D layout
  // (VGPR r, lane L) -> row = t*16 + r + 8*hi, col = 16*wv + lm.
  const int col = wv * 16 + lm;
  const float bv = bias[col];
#pragma unroll
  for (int t = 0; t < 4; ++t) {
#pragma unroll
    for (int r = 0; r < 8; ++r) {
      const int row = r0 + t * 16 + r + 8 * hi;
      float v = acc[t][r] + bv;
      v = v * __frcp_rn(1.0f + __expf(-v));     // SiLU
      if (RES) {
        float rr = resid[row * H + col];
        if (SCALE_RES) rr *= Cv[row / Nn];
        v += rr;
      }
      if (MUL) v *= mulsrc[row * H + col];
      if (OUT_F) outF[row * H + col] = v;
      if (OUT_B) outB[row * H + col] = (__bf16)v;
    }
  }
}

// ---------------------------------------------------------------------------
// Fused edge kernel: per WG of 64 edges,
//   proj = ef[64,32] @ e_w[32,128]  (one WMMA per 16x16 tile, K=32)
//   conv[row[e], :] += scalar_dst[col[e], :] * proj[e, :]   (f32 atomics)
// D layout makes each half-wave touch 16 consecutive columns of one node
// row -> coalesced gathers and coalesced 64B atomic bursts, L2-resident.
// Gathers batched per tile (8 loads in flight); 32-bit unsigned offsets
// keep the saddr + voffset*4 (scale_offset) addressing form.
// ---------------------------------------------------------------------------
__global__ __launch_bounds__(256) void k_edge_scatter(
    const float* __restrict__ ef, const int* __restrict__ eidx,
    const float* __restrict__ sdst, const __bf16* __restrict__ ewt,
    float* __restrict__ conv)
{
  constexpr int SE = 40;   // 80B rows, 16B-aligned chunks
  __shared__ __align__(16) __bf16 sEF[64 * SE];
  __shared__ __align__(16) __bf16 sW[H * SE];
  __shared__ int sRow[64], sCol[64];

  const int tid = threadIdx.x;
  const int e0 = blockIdx.x * 64;              // 64 | E, so one batch per WG
  const int bbase = (e0 / Ee) * Nn;

  if (tid < 64) {
    int2 p = ((const int2*)eidx)[e0 + tid];
    sRow[tid] = p.x;
    sCol[tid] = p.y;
  }
  {
    // Stage ef: 64 rows x 32 f32 -> bf16; 8 floats per thread.
    const int row = tid >> 2, seg = tid & 3;
    const float4* src = (const float4*)ef;
    float4 a = src[(e0 + row) * 8 + seg * 2];
    float4 b = src[(e0 + row) * 8 + seg * 2 + 1];
    H8 p;
    p.h[0] = (__bf16)a.x; p.h[1] = (__bf16)a.y;
    p.h[2] = (__bf16)a.z; p.h[3] = (__bf16)a.w;
    p.h[4] = (__bf16)b.x; p.h[5] = (__bf16)b.y;
    p.h[6] = (__bf16)b.z; p.h[7] = (__bf16)b.w;
    *(uint4*)&sEF[row * SE + seg * 8] = p.u;
  }
  {
    // Stage e_w^T: 128 x 32 bf16 = 512 x 16B, 2 per thread.
    const uint4* src = (const uint4*)ewt;
#pragma unroll
    for (int i = 0; i < 2; ++i) {
      int c = tid + i * 256;
      int wrow = c >> 2, seg = c & 3;
      *(uint4*)&sW[wrow * SE + seg * 8] = src[c];
    }
  }
  __syncthreads();

  const int lane = tid & 31;
  const int wv = tid >> 5;
  const int lm = lane & 15;
  const int hi = lane >> 4;
  const int h = wv * 16 + lm;                  // feature column for this lane
  const unsigned hb = (unsigned)bbase * H + h; // uniform-base lane offset

  AFrag bf;
  bf.u4[0] = *(const uint4*)&sW[h * SE + hi * 8];
  bf.u4[1] = *(const uint4*)&sW[h * SE + 16 + hi * 8];

  const v8f z = {0.f, 0.f, 0.f, 0.f, 0.f, 0.f, 0.f, 0.f};
#pragma unroll
  for (int t = 0; t < 4; ++t) {
    AFrag af;
    const int arow = t * 16 + lm;
    af.u4[0] = *(const uint4*)&sEF[arow * SE + hi * 8];
    af.u4[1] = *(const uint4*)&sEF[arow * SE + 16 + hi * 8];
    v8f acc = __builtin_amdgcn_wmma_f32_16x16x32_bf16(
        false, af.v, false, bf.v, (short)0, z, false, false);

    // Pass 1: pull all 8 edge index pairs for this tile from LDS.
    unsigned gofs[8], cofs[8];
#pragma unroll
    for (int r = 0; r < 8; ++r) {
      const int el = t * 16 + r + 8 * hi;      // edge within block (D layout)
      gofs[r] = hb + (unsigned)sCol[el] * H;
      cofs[r] = hb + (unsigned)sRow[el] * H;
    }
    // Pass 2: issue all 8 gathers back-to-back (8 loads in flight).
    float g[8];
#pragma unroll
    for (int r = 0; r < 8; ++r) g[r] = sdst[gofs[r]];
    // Pass 3: multiply + scatter-add.
#pragma unroll
    for (int r = 0; r < 8; ++r) atomicAdd(&conv[cofs[r]], acc[r] * g[r]);
  }
}

// ---------------------------------------------------------------------------
// Launch: weight prep -> emb MLPs -> fused edge GEMM/scatter -> conv
// residuals (with C scale) -> out residual (with s2s multiply).
// Workspace layout (bytes), ~82.3 MB total:
//   [0)        wt      10*16384 bf16
//   [327680)   ewt     4096 bf16
//   [335872)   bfA     40000*128 bf16 (activation ping)
//   [10575872) bfB     40000*128 bf16 (activation pong)
//   [20815872) s2s     40000*128 f32
//   [41295872) dstb    40000*128 f32 (scalar_dst, then y1)
//   [61775872) convb   40000*128 f32 (conv accum, then x3)
// ---------------------------------------------------------------------------
extern "C" void kernel_launch(void* const* d_in, const int* in_sizes, int n_in,
                              void* d_out, int out_size, void* d_ws, size_t ws_size,
                              hipStream_t stream)
{
  (void)in_sizes; (void)n_in; (void)out_size; (void)ws_size;
  const float* scalar = (const float*)d_in[0];
  const float* ef     = (const float*)d_in[1];
  const int*   eidx   = (const int*)d_in[2];
  const float* Cv     = (const float*)d_in[3];
  const float* emb_w  = (const float*)d_in[4];
  const float* emb_b  = (const float*)d_in[5];
  const float* e_w    = (const float*)d_in[6];
  const float* conv_w = (const float*)d_in[7];
  const float* conv_b = (const float*)d_in[8];
  const float* out_w  = (const float*)d_in[9];
  const float* out_b  = (const float*)d_in[10];
  float* outp = (float*)d_out;

  char* ws = (char*)d_ws;
  __bf16* wt    = (__bf16*)(ws + 0);
  __bf16* ewt   = (__bf16*)(ws + 327680);
  __bf16* bfA   = (__bf16*)(ws + 335872);
  __bf16* bfB   = (__bf16*)(ws + 10575872);
  float*  s2s   = (float*) (ws + 20815872);
  float*  dstb  = (float*) (ws + 41295872);
  float*  convb = (float*) (ws + 61775872);

  const dim3 blk(256);
  const int GG = MT / RPW;          // 625 workgroups per node GEMM

  k_convert_weights<<<656, blk, 0, stream>>>(emb_w, conv_w, out_w, e_w, wt, ewt);
  hipMemsetAsync(convb, 0, (size_t)MT * H * sizeof(float), stream);

  // s2s = Dense(Dense(scalar; emb00); emb01)
  k_gemm_silu<true,false,true,false,false,false,false><<<GG, blk, 0, stream>>>(
      scalar, wt + 0 * 16384, emb_b + 0, nullptr, nullptr, nullptr, nullptr, bfA);
  k_gemm_silu<false,false,false,true,false,false,false><<<GG, blk, 0, stream>>>(
      bfA, wt + 1 * 16384, emb_b + 128, nullptr, nullptr, nullptr, s2s, nullptr);
  // scalar_dst = Dense(Dense(scalar; emb10); emb11)
  k_gemm_silu<true,false,true,false,false,false,false><<<GG, blk, 0, stream>>>(
      scalar, wt + 2 * 16384, emb_b + 256, nullptr, nullptr, nullptr, nullptr, bfA);
  k_gemm_silu<false,false,false,true,false,false,false><<<GG, blk, 0, stream>>>(
      bfA, wt + 3 * 16384, emb_b + 384, nullptr, nullptr, nullptr, dstb, nullptr);

  // conv = segment_sum(gather(scalar_dst, col) * (ef @ e_w), row)
  k_edge_scatter<<<(Bc * Ee) / 64, blk, 0, stream>>>(ef, eidx, dstb, ewt, convb);

  // residual 1 on x = conv*C (C folded into staging/residual reads)
  k_gemm_silu<true,true,true,false,false,false,false><<<GG, blk, 0, stream>>>(
      convb, wt + 4 * 16384, conv_b + 0, Cv, nullptr, nullptr, nullptr, bfA);
  k_gemm_silu<false,false,true,true,true,true,false><<<GG, blk, 0, stream>>>(
      bfA, wt + 5 * 16384, conv_b + 128, Cv, convb, nullptr, dstb, bfB);
  // residual 2 on y1 (in dstb/bfB); fuse x3 = s2s * y2 into epilogue
  k_gemm_silu<false,false,true,false,false,false,false><<<GG, blk, 0, stream>>>(
      bfB, wt + 6 * 16384, conv_b + 256, nullptr, nullptr, nullptr, nullptr, bfA);
  k_gemm_silu<false,false,true,true,true,false,true><<<GG, blk, 0, stream>>>(
      bfA, wt + 7 * 16384, conv_b + 384, nullptr, dstb, s2s, convb, bfB);
  // out residual on x3 (in convb/bfB)
  k_gemm_silu<false,false,true,false,false,false,false><<<GG, blk, 0, stream>>>(
      bfB, wt + 8 * 16384, out_b + 0, nullptr, nullptr, nullptr, nullptr, bfA);
  k_gemm_silu<false,false,false,true,true,false,false><<<GG, blk, 0, stream>>>(
      bfA, wt + 9 * 16384, out_b + 128, nullptr, convb, nullptr, outp, nullptr);
}
